// TransformerEncoder_2078764171893
// MI455X (gfx1250) — matrix operand order repaired and checked by reading, not compiled, not run on hardware
//
#include <hip/hip_runtime.h>
#include <hip/hip_bf16.h>

// MI455X / gfx1250, wave32. Matrix math via v_wmma_f32_16x16x32_bf16
// (bf16 A/B, fp32 accumulate); GEMM tiles staged by the Tensor Data Mover
// (tensor_load_to_lds) with double-buffered LDS, synced via s_wait_tensorcnt.

typedef __bf16 bf16_t;
typedef __attribute__((ext_vector_type(16))) __bf16 v16bf;
typedef __attribute__((ext_vector_type(8)))  __bf16 v8bf;
typedef __attribute__((ext_vector_type(8)))  float  v8f;
typedef __attribute__((ext_vector_type(4)))  unsigned int v4u;
typedef __attribute__((ext_vector_type(8)))  int v8i;
typedef __attribute__((ext_vector_type(4)))  int v4i;

#define BB 4
#define SS 2048
#define EE 1024
#define HH 16
#define DD 64
#define LL 4
static constexpr float kScale = 0.03125f;  // E^-0.5 = 1/32

// ---------------------------------------------------------------- helpers
__device__ __forceinline__ v8bf ld8(const bf16_t* p) { return *(const v8bf*)p; }
__device__ __forceinline__ v16bf cat16(v8bf a, v8bf b) {
  return __builtin_shufflevector(a, b, 0,1,2,3,4,5,6,7,8,9,10,11,12,13,14,15);
}
__device__ __forceinline__ v8f wmma_bf16(v16bf a, v16bf b, v8f c) {
  return __builtin_amdgcn_wmma_f32_16x16x32_bf16(false, a, false, b, (short)0, c,
                                                 false, false);
}

// ---- Tensor Data Mover: 2-D tile load Global->LDS with row padding.
// Descriptor packing per CDNA5 ISA.
// Dims/strides in elements (bf16); pad codes: interval 0=2DW..7=256DW,
// amount 0=1DW..127=128DW.
__device__ __forceinline__ void tdm_load_2d(
    unsigned lds_addr, const void* gptr, unsigned tile_d0, unsigned tile_d1,
    unsigned tensor_d0, unsigned stride_d0, unsigned pad_int_code,
    unsigned pad_amt_code) {
  const unsigned long long ga = (unsigned long long)(size_t)gptr;
  const unsigned tensor_d1 = 1u << 20;  // tiles never exceed bounds

  v4u g0;
  g0[0] = 1u;                                   // count=1, load, user mode
  g0[1] = lds_addr;                             // LDS byte address
  g0[2] = (unsigned)(ga & 0xFFFFFFFFu);         // global_addr[31:0]
  g0[3] = (unsigned)((ga >> 32) & 0x01FFFFFFu)  // global_addr[56:32]
          | (2u << 30);                         // type = 2 ("image")

  v8i g1;
  g1[0] = (int)((1u << 16)                      // data_size = 2 bytes
                | (1u << 20)                    // pad_enable
                | (pad_int_code << 22) | (pad_amt_code << 25));
  g1[1] = (int)((tensor_d0 & 0xFFFFu) << 16);   // atomic_addr=0; dim0 lo16
  g1[2] = (int)(((tensor_d0 >> 16) & 0xFFFFu) | ((tensor_d1 & 0xFFFFu) << 16));
  g1[3] = (int)(((tensor_d1 >> 16) & 0xFFFFu) | (tile_d0 << 16));
  g1[4] = (int)(tile_d1 & 0xFFFFu);             // tile_dim1; tile_dim2 = 0
  g1[5] = (int)stride_d0;                       // dim0 stride [31:0]
  g1[6] = 0;                                    // stride hi / dim1 stride lo
  g1[7] = 0;

  v4i gz = {0, 0, 0, 0};
#if __clang_major__ >= 23
  v8i gz8 = {0, 0, 0, 0, 0, 0, 0, 0};
  __builtin_amdgcn_tensor_load_to_lds(g0, g1, gz, gz, gz8, 0);
#else
  __builtin_amdgcn_tensor_load_to_lds(g0, g1, gz, gz, 0);
#endif
}

// ------------------------------------------------- one-time prep kernels
__global__ __launch_bounds__(256) void transpose_w_bf16(
    const float* __restrict__ W, bf16_t* __restrict__ Wt) {
  size_t idx = (size_t)blockIdx.x * 256 + threadIdx.x;  // l*E*E + n*E + k
  size_t l = idx >> 20;
  size_t n = (idx >> 10) & 1023;
  size_t k = idx & 1023;
  Wt[idx] = (bf16_t)W[(l << 20) + (k << 10) + n];
}

__global__ __launch_bounds__(256) void f32_to_bf16_k(
    const float* __restrict__ in, bf16_t* __restrict__ out, int n) {
  int i = (blockIdx.x * 256 + threadIdx.x) * 4;
  if (i >= n) return;
  float4 f = *(const float4*)(in + i);
  out[i + 0] = (bf16_t)f.x;
  out[i + 1] = (bf16_t)f.y;
  out[i + 2] = (bf16_t)f.z;
  out[i + 3] = (bf16_t)f.w;
}

// 67MB int32 mask -> 2MB bit words so attention never re-reads 4.3GB.
__global__ __launch_bounds__(256) void pack_mask_k(
    const int* __restrict__ mask, unsigned* __restrict__ bits, int nwords) {
  int w = blockIdx.x * 256 + threadIdx.x;
  if (w >= nwords) return;
  const int* p = mask + (size_t)w * 32;
  unsigned acc = 0;
#pragma unroll
  for (int j = 0; j < 32; ++j) acc |= ((unsigned)(p[j] & 1)) << j;
  bits[w] = acc;
}

// ----------------------------------------------------------- bf16 GEMM
// C[M,N] = A[M,K] @ Bt[N,K]^T + bias[N].  256 thr (8 waves), 128x128 tile,
// BK=32, double-buffered LDS filled by TDM (wave 0 issues, all compute).
template <bool OUT_BF16>
__global__ __launch_bounds__(256) void gemm_bf16(
    const bf16_t* __restrict__ A, const bf16_t* __restrict__ Bt,
    const float* __restrict__ bias, bf16_t* __restrict__ Cb,
    float* __restrict__ Cf, int M, int N, int K) {
  __shared__ bf16_t As[2][128 * 40];  // 80B rows (32 elems + 8 pad)
  __shared__ bf16_t Bs[2][128 * 40];

  const int tid = threadIdx.x;
  const int m0 = blockIdx.y * 128;
  const int n0 = blockIdx.x * 128;
  const int wave = tid >> 5, lane = tid & 31;
  const int half = lane >> 4, lm = lane & 15;
  const int wm = wave & 3, wn = wave >> 2;
  const bool issuer = (wave == 0);

  const v8f zero = {0, 0, 0, 0, 0, 0, 0, 0};
  v8f acc[2][4];
#pragma unroll
  for (int i = 0; i < 2; ++i)
#pragma unroll
    for (int j = 0; j < 4; ++j) acc[i][j] = zero;

  // Tile DMA: 32 elems/row (64B = 16 DW, pad code 3), pad 16B (4 DW, code 3),
  // 128 rows, tensor row stride K.
  if (issuer) {
    tdm_load_2d((unsigned)(size_t)&As[0][0], A + (size_t)m0 * K, 32, 128,
                (unsigned)K, (unsigned)K, 3, 3);
    tdm_load_2d((unsigned)(size_t)&Bs[0][0], Bt + (size_t)n0 * K, 32, 128,
                (unsigned)K, (unsigned)K, 3, 3);
    __builtin_amdgcn_s_wait_tensorcnt(0);
  }
  __syncthreads();

  const int nkt = K >> 5;
  for (int kt = 0; kt < nkt; ++kt) {
    const int cur = kt & 1, nxt = cur ^ 1;
    if (issuer && (kt + 1) < nkt) {  // prefetch next tile while computing
      const int k0n = (kt + 1) << 5;
      tdm_load_2d((unsigned)(size_t)&As[nxt][0], A + (size_t)m0 * K + k0n, 32,
                  128, (unsigned)K, (unsigned)K, 3, 3);
      tdm_load_2d((unsigned)(size_t)&Bs[nxt][0], Bt + (size_t)n0 * K + k0n, 32,
                  128, (unsigned)K, (unsigned)K, 3, 3);
    }

    // fragments per ISA wave32 layouts
    v16bf af[2];
#pragma unroll
    for (int sm = 0; sm < 2; ++sm) {
      const bf16_t* p = &As[cur][(wm * 32 + sm * 16 + lm) * 40 + half * 8];
      af[sm] = cat16(ld8(p), ld8(p + 16));  // A: e<8 -> k0+8*half+e, e>=8 -> k0+16+8*half+(e-8)
    }
    v16bf bfr[4];
#pragma unroll
    for (int sn = 0; sn < 4; ++sn) {
      const bf16_t* p = &Bs[cur][(wn * 64 + sn * 16 + lm) * 40 + half * 8];
      bfr[sn] = cat16(ld8(p), ld8(p + 16));  // B: e<8 -> k0+8*half+e, e>=8 -> k0+16+8*half+(e-8), n = lm
    }
#pragma unroll
    for (int sm = 0; sm < 2; ++sm)
#pragma unroll
      for (int sn = 0; sn < 4; ++sn)
        acc[sm][sn] = wmma_bf16(af[sm], bfr[sn], acc[sm][sn]);

    if (issuer) __builtin_amdgcn_s_wait_tensorcnt(0);
    __syncthreads();
  }

  // epilogue: C layout M = r + 8*half, N = lm
#pragma unroll
  for (int sm = 0; sm < 2; ++sm)
#pragma unroll
    for (int sn = 0; sn < 4; ++sn) {
      int n = n0 + wn * 64 + sn * 16 + lm;
      float bs = bias[n];
#pragma unroll
      for (int r = 0; r < 8; ++r) {
        int m = m0 + wm * 32 + sm * 16 + r + 8 * half;
        float val = acc[sm][sn][r] + bs;
        if constexpr (OUT_BF16)
          Cb[(size_t)m * N + n] = (bf16_t)val;
        else
          Cf[(size_t)m * N + n] = val;
      }
    }
}

// ----------------------------------------------------- ReLU attention
// No softmax => streaming accumulation. Block: 256 thr, one (b,h),
// 128 query rows (16/wave), 32-key blocks. K tile staged by TDM,
// V tile transposed manually (TDM cannot transpose).
__global__ __launch_bounds__(256) void relu_attention(
    const bf16_t* __restrict__ q, const bf16_t* __restrict__ k,
    const bf16_t* __restrict__ v, const unsigned* __restrict__ mbits,
    bf16_t* __restrict__ o) {
  __shared__ bf16_t Ks[32 * 72];     // [key][d], 144B rows (64 + 8 pad)
  __shared__ bf16_t Vt[64 * 40];     // [d][key], transposed
  __shared__ bf16_t Aw[8][16 * 40];  // per-wave attn tile re-layout scratch

  const int tid = threadIdx.x;
  const int b = blockIdx.z, h = blockIdx.y;
  const int q0 = blockIdx.x * 128;
  const int wave = tid >> 5, lane = tid & 31;
  const int half = lane >> 4, lm = lane & 15;
  const int qw = q0 + wave * 16;

  const bf16_t* qrow = q + ((size_t)(b * SS + qw + lm)) * EE + h * DD;
  v16bf aq[2];
#pragma unroll
  for (int c = 0; c < 2; ++c) {
    const bf16_t* p = qrow + c * 32 + half * 8;
    aq[c] = cat16(ld8(p), ld8(p + 16));
  }

  const v8f zero = {0, 0, 0, 0, 0, 0, 0, 0};
  v8f oacc[4];
#pragma unroll
  for (int i = 0; i < 4; ++i) oacc[i] = zero;

  for (int kb = 0; kb < SS; kb += 32) {
    // V: manual transposed stage. K: TDM 2-D tile (64 elem rows = 32 DW,
    // pad code 4; +16B pad = 4 DW, code 3; row stride E).
    {
      int row = tid >> 3, cc = tid & 7;
      size_t src = ((size_t)(b * SS + kb + row)) * EE + h * DD + cc * 8;
      v8bf vv = ld8(v + src);
#pragma unroll
      for (int j = 0; j < 8; ++j) Vt[(cc * 8 + j) * 40 + row] = vv[j];
    }
    if (wave == 0) {
      tdm_load_2d((unsigned)(size_t)&Ks[0],
                  k + ((size_t)(b * SS + kb)) * EE + h * DD, 64, 32, 64,
                  (unsigned)EE, 4, 3);
      __builtin_amdgcn_s_wait_tensorcnt(0);
    }
    __syncthreads();

    unsigned mw[8];
    const int kw = kb >> 5;
#pragma unroll
    for (int r = 0; r < 8; ++r)
      mw[r] = mbits[((size_t)(b * SS + qw + r + 8 * half)) * (SS / 32) + kw];

    // energy = q @ k^T per 16-key subtile (2 chained WMMAs over d=64)
#pragma unroll
    for (int sub = 0; sub < 2; ++sub) {
      v8f e = zero;
#pragma unroll
      for (int c = 0; c < 2; ++c) {
        const bf16_t* p = &Ks[(sub * 16 + lm) * 72 + c * 32 + half * 8];
        v16bf bk = cat16(ld8(p), ld8(p + 16));
        e = wmma_bf16(aq[c], bk, e);
      }
      // mask==1 -> 0 (ref: energy=-1e-8 -> relu -> 0); else relu(e*scale)
#pragma unroll
      for (int r = 0; r < 8; ++r) {
        float a = e[r] * kScale;
        a = ((mw[r] >> (sub * 16 + lm)) & 1u) ? 0.f : fmaxf(a, 0.f);
        Aw[wave][(r + 8 * half) * 40 + sub * 16 + lm] = (bf16_t)a;
      }
    }

    // o += attn @ V (A-frag re-read from per-wave LDS scratch)
    const bf16_t* pa = &Aw[wave][lm * 40 + half * 8];
    v16bf aa = cat16(ld8(pa), ld8(pa + 16));
#pragma unroll
    for (int sd = 0; sd < 4; ++sd) {
      const bf16_t* p = &Vt[(sd * 16 + lm) * 40 + half * 8];
      v16bf bv = cat16(ld8(p), ld8(p + 16));
      oacc[sd] = wmma_bf16(aa, bv, oacc[sd]);
    }
    __syncthreads();
  }

#pragma unroll
  for (int sd = 0; sd < 4; ++sd)
#pragma unroll
    for (int r = 0; r < 8; ++r) {
      int m = qw + r + 8 * half;
      o[((size_t)(b * SS + m)) * EE + h * DD + sd * 16 + lm] =
          (bf16_t)oacc[sd][r];
    }
}

// --------------------------------------------- residual + LayerNorm
__global__ __launch_bounds__(256) void add_layernorm(
    const float* __restrict__ oproj, const float* __restrict__ xres,
    const float* __restrict__ gamma, const float* __restrict__ beta,
    float* __restrict__ xout, bf16_t* __restrict__ xbf) {
  const int row = blockIdx.x, t = threadIdx.x;
  const size_t base = (size_t)row * EE + t * 4;
  float4 a = *(const float4*)(oproj + base);
  float4 bsrc = *(const float4*)(xres + base);
  float v0 = a.x + bsrc.x, v1 = a.y + bsrc.y, v2 = a.z + bsrc.z,
        v3 = a.w + bsrc.w;
  float s = v0 + v1 + v2 + v3;
  float ss = v0 * v0 + v1 * v1 + v2 * v2 + v3 * v3;
#pragma unroll
  for (int off = 16; off > 0; off >>= 1) {
    s += __shfl_down(s, off);
    ss += __shfl_down(ss, off);
  }
  __shared__ float rs[8], rss[8];
  const int wave = t >> 5, lane = t & 31;
  if (lane == 0) { rs[wave] = s; rss[wave] = ss; }
  __syncthreads();
  if (t == 0) {
    float S1 = 0.f, S2 = 0.f;
#pragma unroll
    for (int i = 0; i < 8; ++i) { S1 += rs[i]; S2 += rss[i]; }
    rs[0] = S1; rss[0] = S2;
  }
  __syncthreads();
  const float mean = rs[0] * (1.f / EE);
  const float var = rss[0] * (1.f / EE) - mean * mean;
  const float rstd = rsqrtf(var + 1e-5f);
  const int col = t * 4;
  float4 g = *(const float4*)(gamma + col);
  float4 be = *(const float4*)(beta + col);
  float o0 = (v0 - mean) * rstd * g.x + be.x;
  float o1 = (v1 - mean) * rstd * g.y + be.y;
  float o2 = (v2 - mean) * rstd * g.z + be.z;
  float o3 = (v3 - mean) * rstd * g.w + be.w;
  float4 ov = {o0, o1, o2, o3};
  *(float4*)(xout + base) = ov;
  xbf[base + 0] = (bf16_t)o0;
  xbf[base + 1] = (bf16_t)o1;
  xbf[base + 2] = (bf16_t)o2;
  xbf[base + 3] = (bf16_t)o3;
}

// ------------------------------------------------------------- launch
extern "C" void kernel_launch(void* const* d_in, const int* in_sizes, int n_in,
                              void* d_out, int out_size, void* d_ws,
                              size_t ws_size, hipStream_t stream) {
  (void)in_sizes; (void)n_in; (void)out_size; (void)ws_size;

  const float* query = (const float*)d_in[0];
  const float* value = (const float*)d_in[1];
  const int*   mask  = (const int*)d_in[2];
  const float* Wq = (const float*)d_in[3];
  const float* Wk = (const float*)d_in[4];
  const float* Wv = (const float*)d_in[5];
  const float* Wo = (const float*)d_in[6];
  const float* bq = (const float*)d_in[7];
  const float* bk = (const float*)d_in[8];
  const float* bv = (const float*)d_in[9];
  const float* bo = (const float*)d_in[10];
  const float* gamma = (const float*)d_in[11];
  const float* beta  = (const float*)d_in[12];
  float* out = (float*)d_out;

  constexpr size_t WB  = (size_t)LL * EE * EE * sizeof(bf16_t);   // 8 MiB
  constexpr size_t XBF = (size_t)BB * SS * EE * sizeof(bf16_t);   // 16 MiB
  constexpr size_t XF  = (size_t)BB * SS * EE * sizeof(float);    // 32 MiB
  constexpr int NW = BB * SS * SS / 32;                           // mask words

  char* w = (char*)d_ws;
  bf16_t* WqT = (bf16_t*)(w);
  bf16_t* WkT = (bf16_t*)(w + 1 * WB);
  bf16_t* WvT = (bf16_t*)(w + 2 * WB);
  bf16_t* WoT = (bf16_t*)(w + 3 * WB);
  bf16_t* valbf = (bf16_t*)(w + 4 * WB);
  bf16_t* xbf   = (bf16_t*)(w + 4 * WB + 1 * XBF);
  bf16_t* qbf   = (bf16_t*)(w + 4 * WB + 2 * XBF);
  bf16_t* kbf   = (bf16_t*)(w + 4 * WB + 3 * XBF);
  bf16_t* vbf   = (bf16_t*)(w + 4 * WB + 4 * XBF);
  bf16_t* obf   = (bf16_t*)(w + 4 * WB + 5 * XBF);
  float*  oproj = (float*)(w + 4 * WB + 6 * XBF);
  float*  xf    = (float*)(w + 4 * WB + 6 * XBF + XF);
  unsigned* mbits = (unsigned*)(w + 4 * WB + 6 * XBF + 2 * XF);

  const int M = BB * SS, N = EE, K = EE;
  const int nElem = BB * SS * EE;

  const int twBlocks = (LL * EE * EE) / 256;
  transpose_w_bf16<<<twBlocks, 256, 0, stream>>>(Wq, WqT);
  transpose_w_bf16<<<twBlocks, 256, 0, stream>>>(Wk, WkT);
  transpose_w_bf16<<<twBlocks, 256, 0, stream>>>(Wv, WvT);
  transpose_w_bf16<<<twBlocks, 256, 0, stream>>>(Wo, WoT);
  f32_to_bf16_k<<<(nElem / 4 + 255) / 256, 256, 0, stream>>>(value, valbf, nElem);
  f32_to_bf16_k<<<(nElem / 4 + 255) / 256, 256, 0, stream>>>(query, xbf, nElem);
  pack_mask_k<<<(NW + 255) / 256, 256, 0, stream>>>(mask, mbits, NW);

  const dim3 gGemm(N / 128, M / 128);
  const dim3 gAttn(SS / 128, HH, BB);

  for (int l = 0; l < LL; ++l) {
    const size_t wo = (size_t)l * EE * EE;
    gemm_bf16<true><<<gGemm, 256, 0, stream>>>(xbf, WqT + wo, bq + l * EE,
                                               qbf, nullptr, M, N, K);
    gemm_bf16<true><<<gGemm, 256, 0, stream>>>(valbf, WkT + wo, bk + l * EE,
                                               kbf, nullptr, M, N, K);
    gemm_bf16<true><<<gGemm, 256, 0, stream>>>(valbf, WvT + wo, bv + l * EE,
                                               vbf, nullptr, M, N, K);
    relu_attention<<<gAttn, 256, 0, stream>>>(qbf, kbf, vbf, mbits, obf);
    gemm_bf16<false><<<gGemm, 256, 0, stream>>>(obf, WoT + wo, bo + l * EE,
                                                nullptr, oproj, M, N, K);
    const float* xres = (l == 0) ? query : xf;
    float* xo = (l == LL - 1) ? out : xf;
    add_layernorm<<<M, 256, 0, stream>>>(oproj, xres, gamma + l * EE,
                                         beta + l * EE, xo, xbf);
  }
}
